// GlobalAdaptiveTSM_47115791237652
// MI455X (gfx1250) — matrix operand order repaired and checked by reading, not compiled, not run on hardware
//
#include <hip/hip_runtime.h>

typedef unsigned short u16;
typedef __attribute__((ext_vector_type(16))) __bf16 v16bf;
typedef __attribute__((ext_vector_type(8)))  __bf16 v8bf;
typedef __attribute__((ext_vector_type(8)))  float  v8f;

static constexpr int kB = 16, kT = 512, kD = 512, kH = 4, kHD = 128, kFF = 2048;
static constexpr int kBT = kB * kT;

__device__ __forceinline__ u16 f2bf(float f) {
  unsigned u = __float_as_uint(f);
  u += 0x7FFFu + ((u >> 16) & 1u);
  return (u16)(u >> 16);
}

__device__ __forceinline__ float wsum(float v) {
#pragma unroll
  for (int o = 16; o > 0; o >>= 1) v += __shfl_xor(v, o, 32);
  return v;
}
__device__ __forceinline__ float wmaxr(float v) {
#pragma unroll
  for (int o = 16; o > 0; o >>= 1) v = fmaxf(v, __shfl_xor(v, o, 32));
  return v;
}

// ---- CDNA5 async global->LDS DMA (ASYNCcnt-tracked), 16B per lane ----
__device__ __forceinline__ void async_b128(const u16* gptr, u16* lptr) {
  unsigned lds = (unsigned)(unsigned long long)(const void*)lptr; // aperture: low 32 bits == LDS offset
  asm volatile("global_load_async_to_lds_b128 %0, %1, off"
               :: "v"(lds), "v"(gptr) : "memory");
}
__device__ __forceinline__ void wait_async0() {
  asm volatile("s_wait_asynccnt 0x0" ::: "memory");
}
__device__ __forceinline__ void wait_async2() {
  asm volatile("s_wait_asynccnt 0x2" ::: "memory");
}

// ---------------- elementwise f32 -> bf16 (weights) ----------------
__global__ void k_f32_to_bf16(const float* __restrict__ s, u16* __restrict__ d, int n) {
  int i = blockIdx.x * blockDim.x + threadIdx.x;
  if (i < n) d[i] = f2bf(s[i]);
}

// ---------------- adaptive shift: one wave per token ----------------
__global__ void k_shift(const float* __restrict__ x, const float* __restrict__ Wp,
                        const float* __restrict__ bp, float* __restrict__ of,
                        u16* __restrict__ ob) {
  int tok  = (int)((blockIdx.x * blockDim.x + threadIdx.x) >> 5);
  int lane = threadIdx.x & 31;
  int t = tok % kT;
  const float* row = x + (size_t)tok * kD;
  float s = 0.f;
#pragma unroll
  for (int j = 0; j < 16; ++j) s += row[lane * 16 + j] * Wp[lane * 16 + j];
  s = wsum(s);
  float ratio = 1.f / (1.f + __expf(-(s + bp[0])));
  int shn = (int)floorf(ratio * (float)(kD / 2));
#pragma unroll
  for (int j = 0; j < 16; ++j) {
    int c = j * 32 + lane;
    float v;
    if (c < shn)          v = (t > 0)      ? row[c - kD] : 0.f;   // from t-1
    else if (c < 2 * shn) v = (t < kT - 1) ? row[c + kD] : 0.f;   // from t+1
    else                  v = row[c];
    of[(size_t)tok * kD + c] = v;
    ob[(size_t)tok * kD + c] = f2bf(v);
  }
}

// ---------------- generic NT bf16 WMMA GEMM ----------------
// C[m,n] = scale * sum_k A[m,k]*Bn[n,k] + bias[n]; optional ReLU; f32 and/or bf16 out.
// Batched via blockIdx.z: off = (z/Hdiv)*s?b + (z%Hdiv)*s?h (element offsets).
// Block = 128 threads (4 waves). Block tile 128(M) x 64(N); wave tile 32 x 64.
// B tile (64n x 32k bf16, 4KB) staged in LDS via async DMA, double-buffered.
__global__ __launch_bounds__(128)
void k_gemm_nt(const u16* __restrict__ A, int lda, long long sAb, long long sAh,
               const u16* __restrict__ Bn, int ldb, long long sBb, long long sBh,
               float* __restrict__ Cf, u16* __restrict__ Cb, int ldc,
               long long sCb, long long sCh,
               const float* __restrict__ bias,
               int Kd, int Hdiv, float scale, int relu) {
  __shared__ u16 Bt[2][64][32];   // [buf][n][k] : 8 KB

  int z  = blockIdx.z;
  int zb = z / Hdiv, zh = z % Hdiv;
  const u16* Ab = A + zb * sAb + zh * sAh;
  const u16* Bb = Bn + zb * sBb + zh * sBh;
  long long offC = zb * sCb + zh * sCh;

  int tid  = threadIdx.x;
  int lane = tid & 31;
  int wv   = tid >> 5;
  int half = lane >> 4, l15 = lane & 15;
  int m0 = blockIdx.x * 128 + wv * 32;   // wave covers rows [m0, m0+32)
  int n0 = blockIdx.y * 64;

  // fill assignment: 128 threads x 32B each = 4KB tile
  int fr = tid >> 1;            // row 0..63
  int fe = (tid & 1) * 16;      // element 0 or 16 within the 32-elem row

  v8f zero8 = {0.f, 0.f, 0.f, 0.f, 0.f, 0.f, 0.f, 0.f};
  v8f acc[2][4];
#pragma unroll
  for (int mi = 0; mi < 2; ++mi)
#pragma unroll
    for (int j = 0; j < 4; ++j) acc[mi][j] = zero8;

  union F16 { v16bf v; v8bf h[2]; };

  const u16* frow = Bb + (size_t)(n0 + fr) * ldb + fe;  // + kk each step
  // prologue fill of buffer 0
  {
    u16* l = &Bt[0][fr][fe];
    async_b128(frow, l);
    async_b128(frow + 8, l + 8);
  }

  const u16* arow0 = Ab + (size_t)(m0 + l15) * lda;
  const u16* arow1 = arow0 + (size_t)16 * lda;

  for (int kk = 0; kk < Kd; kk += 32) {
    int cur = (kk >> 5) & 1;
    bool more = (kk + 32) < Kd;
    if (more) {  // issue prefetch of next tile (uniform branch)
      const u16* g = frow + kk + 32;
      u16* l = &Bt[cur ^ 1][fr][fe];
      async_b128(g, l);
      async_b128(g + 8, l + 8);
      wait_async2();             // current tile's 2 DMAs done; 2 prefetch in flight
    } else {
      wait_async0();
    }
    __syncthreads();             // all waves' fills of `cur` visible

    F16 a0, a1;
    const u16* ap0 = arow0 + kk + half * 8;
    a0.h[0] = *reinterpret_cast<const v8bf*>(ap0);
    a0.h[1] = *reinterpret_cast<const v8bf*>(ap0 + 16);
    const u16* ap1 = arow1 + kk + half * 8;
    a1.h[0] = *reinterpret_cast<const v8bf*>(ap1);
    a1.h[1] = *reinterpret_cast<const v8bf*>(ap1 + 16);

#pragma unroll
    for (int j = 0; j < 4; ++j) {
      F16 bf_;
      const u16* bl = &Bt[cur][j * 16 + l15][half * 8];
      bf_.h[0] = *reinterpret_cast<const v8bf*>(bl);      // ds_load_b128
      bf_.h[1] = *reinterpret_cast<const v8bf*>(bl + 16);
      acc[0][j] = __builtin_amdgcn_wmma_f32_16x16x32_bf16(
          false, a0.v, false, bf_.v, (short)0, acc[0][j], false, false);
      acc[1][j] = __builtin_amdgcn_wmma_f32_16x16x32_bf16(
          false, a1.v, false, bf_.v, (short)0, acc[1][j], false, false);
    }
    __syncthreads();             // protect `cur` (== buf refilled by step kk+32's prefetch) from WAR
  }

#pragma unroll
  for (int j = 0; j < 4; ++j) {
    int col = n0 + j * 16 + l15;
    float bv = bias ? bias[col] : 0.f;
#pragma unroll
    for (int mi = 0; mi < 2; ++mi) {
#pragma unroll
      for (int r = 0; r < 8; ++r) {
        int row = m0 + mi * 16 + half * 8 + r;
        float v = acc[mi][j][r] * scale + bv;
        if (relu) v = fmaxf(v, 0.f);
        size_t idx = (size_t)offC + (size_t)row * ldc + col;
        if (Cf) Cf[idx] = v;
        if (Cb) Cb[idx] = f2bf(v);
      }
    }
  }
}

// ---------------- V transpose: vt[b,h,d,t] = qkv[b,t, 1024 + h*128 + d] ----------------
__global__ void k_vt(const u16* __restrict__ qkv, u16* __restrict__ vt) {
  size_t i = (size_t)blockIdx.x * blockDim.x + threadIdx.x;
  if (i >= (size_t)kB * kH * kHD * kT) return;
  int t = (int)(i % kT); size_t r = i / kT;
  int d = (int)(r % kHD); r /= kHD;
  int h = (int)(r % kH);  int b = (int)(r / kH);
  vt[i] = qkv[((size_t)(b * kT + t)) * (3 * kD) + 2 * kD + h * kHD + d];
}

// ---------------- softmax: one wave per row of 512, f32 in, bf16 out ----------------
__global__ void k_softmax(const float* __restrict__ s, u16* __restrict__ a) {
  int row  = (int)((blockIdx.x * blockDim.x + threadIdx.x) >> 5);
  int lane = threadIdx.x & 31;
  const float* sr = s + (size_t)row * kT;
  float v[16]; float mx = -1e30f;
#pragma unroll
  for (int j = 0; j < 16; ++j) { v[j] = sr[j * 32 + lane]; mx = fmaxf(mx, v[j]); }
  mx = wmaxr(mx);
  float sum = 0.f;
#pragma unroll
  for (int j = 0; j < 16; ++j) { v[j] = __expf(v[j] - mx); sum += v[j]; }
  sum = wsum(sum);
  float inv = 1.f / sum;
#pragma unroll
  for (int j = 0; j < 16; ++j) a[(size_t)row * kT + j * 32 + lane] = f2bf(v[j] * inv);
}

// ---------------- residual add + LayerNorm: one wave per token ----------------
__global__ void k_add_ln(const float* __restrict__ res, const float* __restrict__ add,
                         const float* __restrict__ g, const float* __restrict__ bta,
                         float* __restrict__ of, u16* __restrict__ ob) {
  int tok  = (int)((blockIdx.x * blockDim.x + threadIdx.x) >> 5);
  int lane = threadIdx.x & 31;
  size_t base = (size_t)tok * kD;
  float y[16]; float s = 0.f, sq = 0.f;
#pragma unroll
  for (int j = 0; j < 16; ++j) {
    int c = j * 32 + lane;
    float v = res[base + c] + add[base + c];
    y[j] = v; s += v; sq += v * v;
  }
  s = wsum(s); sq = wsum(sq);
  float mean = s * (1.f / kD);
  float var  = sq * (1.f / kD) - mean * mean;
  float rs   = rsqrtf(var + 1e-5f);
#pragma unroll
  for (int j = 0; j < 16; ++j) {
    int c = j * 32 + lane;
    float o = (y[j] - mean) * rs * g[c] + bta[c];
    of[base + c] = o;
    if (ob) ob[base + c] = f2bf(o);
  }
}

extern "C" void kernel_launch(void* const* d_in, const int* in_sizes, int n_in,
                              void* d_out, int out_size, void* d_ws, size_t ws_size,
                              hipStream_t stream) {
  (void)in_sizes; (void)n_in; (void)out_size; (void)ws_size;
  const float* x    = (const float*)d_in[0];
  const float* Wp   = (const float*)d_in[1];
  const float* bp   = (const float*)d_in[2];
  const float* Wqkv = (const float*)d_in[3];
  const float* bqkv = (const float*)d_in[4];
  const float* Wo   = (const float*)d_in[5];
  const float* bo   = (const float*)d_in[6];
  const float* W1   = (const float*)d_in[7];
  const float* b1   = (const float*)d_in[8];
  const float* W2   = (const float*)d_in[9];
  const float* b2   = (const float*)d_in[10];
  const float* ln1s = (const float*)d_in[11];
  const float* ln1b = (const float*)d_in[12];
  const float* ln2s = (const float*)d_in[13];
  const float* ln2b = (const float*)d_in[14];

  char* wsb = (char*)d_ws; size_t off = 0;
  auto bump = [&](size_t bytes) -> void* {
    void* p = wsb + off; off = (off + bytes + 255) & ~(size_t)255; return p;
  };

  u16*   wqkv_bf = (u16*)bump((size_t)2 * 3 * kD * kD * 2);
  u16*   wo_bf   = (u16*)bump((size_t)2 * kD * kD * 2);
  u16*   w1_bf   = (u16*)bump((size_t)2 * kFF * kD * 2);
  u16*   w2_bf   = (u16*)bump((size_t)2 * kD * kFF * 2);
  float* xA_f    = (float*)bump((size_t)kBT * kD * 4);
  u16*   xA_b    = (u16*)bump((size_t)kBT * kD * 2);
  float* xB_f    = (float*)bump((size_t)kBT * kD * 4);
  u16*   xB_b    = (u16*)bump((size_t)kBT * kD * 2);
  u16*   qkv_b   = (u16*)bump((size_t)kBT * 3 * kD * 2);
  u16*   vt_b    = (u16*)bump((size_t)kB * kH * kHD * kT * 2);
  float* sc_f    = (float*)bump((size_t)kB * kH * kT * kT * 4);
  u16*   at_b    = (u16*)bump((size_t)kB * kH * kT * kT * 2);
  u16*   ctx_b   = (u16*)bump((size_t)kBT * kD * 2);
  float* tmp_f   = (float*)bump((size_t)kBT * kD * 4);
  u16*   hid_b   = (u16*)bump((size_t)kBT * kFF * 2);

  // ---- weights to bf16 ----
  int n;
  n = 2 * 3 * kD * kD; k_f32_to_bf16<<<dim3((n + 255) / 256), dim3(256), 0, stream>>>(Wqkv, wqkv_bf, n);
  n = 2 * kD * kD;     k_f32_to_bf16<<<dim3((n + 255) / 256), dim3(256), 0, stream>>>(Wo,   wo_bf,   n);
  n = 2 * kFF * kD;    k_f32_to_bf16<<<dim3((n + 255) / 256), dim3(256), 0, stream>>>(W1,   w1_bf,   n);
  n = 2 * kD * kFF;    k_f32_to_bf16<<<dim3((n + 255) / 256), dim3(256), 0, stream>>>(W2,   w2_bf,   n);

  // ---- adaptive shift ----
  k_shift<<<dim3(kBT / 8), dim3(256), 0, stream>>>(x, Wp, bp, xA_f, xA_b);

  auto gemm = [&](const u16* A, int lda, long long sAb, long long sAh,
                  const u16* Bn, int ldb, long long sBb, long long sBh,
                  float* Cf, u16* Cb, int ldc, long long sCb, long long sCh,
                  const float* bias, int M, int N, int Kd, int Z, int Hdiv,
                  float scale, int relu) {
    dim3 g((unsigned)(M / 128), (unsigned)(N / 64), (unsigned)Z);
    k_gemm_nt<<<g, dim3(128), 0, stream>>>(A, lda, sAb, sAh, Bn, ldb, sBb, sBh,
                                           Cf, Cb, ldc, sCb, sCh, bias, Kd, Hdiv,
                                           scale, relu);
  };

  const float scl = 0.0883883476483184f; // 1/sqrt(128)
  for (int L = 0; L < 2; ++L) {
    // QKV projection: [8192,512] x [1536,512]^T -> bf16 [8192,1536]
    gemm(xA_b, kD, 0, 0,
         wqkv_bf + (size_t)L * 3 * kD * kD, kD, 0, 0,
         nullptr, qkv_b, 3 * kD, 0, 0,
         bqkv + (size_t)L * 3 * kD, kBT, 3 * kD, kD, 1, 1, 1.f, 0);

    // V transpose
    n = kB * kH * kHD * kT;
    k_vt<<<dim3((n + 255) / 256), dim3(256), 0, stream>>>(qkv_b, vt_b);

    // scores = (Q K^T)/sqrt(HD), batched over b*H (64)
    gemm(qkv_b, 3 * kD, (long long)kT * 3 * kD, kHD,
         qkv_b + kD, 3 * kD, (long long)kT * 3 * kD, kHD,
         sc_f, nullptr, kT, (long long)kH * kT * kT, (long long)kT * kT,
         nullptr, kT, kT, kHD, kB * kH, kH, scl, 0);

    // softmax -> bf16 attn
    k_softmax<<<dim3(kB * kH * kT / 8), dim3(256), 0, stream>>>(sc_f, at_b);

    // ctx = attn @ V  (V^T stored [HD,T]) -> bf16 [8192,512] (heads interleaved)
    gemm(at_b, kT, (long long)kH * kT * kT, (long long)kT * kT,
         vt_b, kT, (long long)kH * kHD * kT, (long long)kHD * kT,
         nullptr, ctx_b, kD, (long long)kT * kD, (long long)kHD,
         nullptr, kT, kHD, kT, kB * kH, kH, 1.f, 0);

    // output projection -> f32
    gemm(ctx_b, kD, 0, 0,
         wo_bf + (size_t)L * kD * kD, kD, 0, 0,
         tmp_f, nullptr, kD, 0, 0,
         bo + (size_t)L * kD, kBT, kD, kD, 1, 1, 1.f, 0);

    // x1 = LN(x + attn_out)
    k_add_ln<<<dim3(kBT / 8), dim3(256), 0, stream>>>(
        xA_f, tmp_f, ln1s + (size_t)L * kD, ln1b + (size_t)L * kD, xB_f, xB_b);

    // FF1 with fused ReLU -> bf16 [8192,2048]
    gemm(xB_b, kD, 0, 0,
         w1_bf + (size_t)L * kFF * kD, kD, 0, 0,
         nullptr, hid_b, kFF, 0, 0,
         b1 + (size_t)L * kFF, kBT, kFF, kD, 1, 1, 1.f, 1);

    // FF2 -> f32
    gemm(hid_b, kFF, 0, 0,
         w2_bf + (size_t)L * kD * kFF, kFF, 0, 0,
         tmp_f, nullptr, kD, 0, 0,
         b2 + (size_t)L * kD, kBT, kD, kFF, 1, 1, 1.f, 0);

    // x = LN(x1 + ff); final layer writes f32 straight to d_out
    float* outf = (L == 1) ? (float*)d_out : xA_f;
    k_add_ln<<<dim3(kBT / 8), dim3(256), 0, stream>>>(
        xB_f, tmp_f, ln2s + (size_t)L * kD, ln2b + (size_t)L * kD, outf, xA_b);
  }
}
